// Attention_30709016167214
// MI455X (gfx1250) — matrix operand order repaired and checked by reading, not compiled, not run on hardware
//
#include <hip/hip_runtime.h>

typedef unsigned short u16t;
typedef unsigned int   u32t;
typedef __attribute__((ext_vector_type(16))) __bf16 v16bf;
typedef __attribute__((ext_vector_type(8)))  float  v8f;

struct U8x32 { uint4 x; uint4 y; };   // 32 bytes == v16bf

__device__ __forceinline__ u16t f2bf(float f) {
  u32t u = __builtin_bit_cast(u32t, f);
  u32t r = u + 0x7fffu + ((u >> 16) & 1u);   // round-to-nearest-even
  return (u16t)(r >> 16);
}

// Build a v16bf fragment from two 16-byte aligned chunks (LDS or global).
__device__ __forceinline__ v16bf ld_frag(const u16t* p0, const u16t* p1) {
  U8x32 t;
  t.x = *(const uint4*)p0;
  t.y = *(const uint4*)p1;
  return __builtin_bit_cast(v16bf, t);
}

// ---------------------------------------------------------------------------
// fp32 -> bf16 conversion (2 elements / thread, packed dword stores)
// ---------------------------------------------------------------------------
__global__ void cvt_f32_bf16(const float* __restrict__ src,
                             u16t* __restrict__ dst, int npairs) {
  int i = blockIdx.x * blockDim.x + threadIdx.x;
  if (i < npairs) {
    float2 f = ((const float2*)src)[i];
    ((u32t*)dst)[i] = (u32t)f2bf(f.x) | ((u32t)f2bf(f.y) << 16);
  }
}

// ---------------------------------------------------------------------------
// bf16 GEMM:  C[m,n] = sum_k X[m,k] * W[n,k]   (X: 8192x1024, W: 1024x1024)
// mode 0: store bf16 into [B,H,L,DH] layout (for q/k/v)
// mode 1: store fp32 row-major [m,n] (final output)
// Workgroup: 256 threads = 8 waves (4x2); tile 128x128; wave tile 32x64;
// K-step 32; double-buffered LDS, one barrier per K-step.
// ---------------------------------------------------------------------------
__global__ __launch_bounds__(256) void gemm_bf16(const u16t* __restrict__ X,
                                                 const u16t* __restrict__ W,
                                                 void* __restrict__ out,
                                                 int mode) {
  __shared__ u16t Xl[2][128][40];   // 80B row stride -> aligned b128 frags
  __shared__ u16t Wl[2][128][40];

  const int tid  = threadIdx.x;
  const int wave = tid >> 5, lane = tid & 31;
  const int lh   = lane >> 4, ln = lane & 15;
  const int wm   = wave >> 1, wn = wave & 1;     // 4x2 wave grid
  const int gm   = blockIdx.x * 128, gn = blockIdx.y * 128;

  v8f acc[2][4];
#pragma unroll
  for (int mt = 0; mt < 2; ++mt)
#pragma unroll
    for (int nt = 0; nt < 4; ++nt)
#pragma unroll
      for (int i = 0; i < 8; ++i) acc[mt][nt][i] = 0.0f;

  // Loader: each thread moves 2 uint4 per matrix per K-step.
  const int lr  = tid >> 1;          // 0..127 (row)
  const int lc0 = (tid & 1) * 2;     // uint4 index within 32-elem row chunk

  const u16t* Xg = X + (size_t)(gm + lr) * 1024 + lc0 * 8;
  const u16t* Wg = W + (size_t)(gn + lr) * 1024 + lc0 * 8;

  uint4 px0, px1, pw0, pw1;
  px0 = *(const uint4*)(Xg);      px1 = *(const uint4*)(Xg + 8);
  pw0 = *(const uint4*)(Wg);      pw1 = *(const uint4*)(Wg + 8);
  *(uint4*)&Xl[0][lr][lc0 * 8]     = px0;
  *(uint4*)&Xl[0][lr][lc0 * 8 + 8] = px1;
  *(uint4*)&Wl[0][lr][lc0 * 8]     = pw0;
  *(uint4*)&Wl[0][lr][lc0 * 8 + 8] = pw1;
  __syncthreads();

  for (int t = 0; t < 32; ++t) {
    const int cur = t & 1, nxt = cur ^ 1;
    if (t < 31) {                      // prefetch next K-slice into registers
      const int k0 = (t + 1) * 32;
      px0 = *(const uint4*)(Xg + k0);  px1 = *(const uint4*)(Xg + k0 + 8);
      pw0 = *(const uint4*)(Wg + k0);  pw1 = *(const uint4*)(Wg + k0 + 8);
    }

    // A fragments: lane ln = row M; elems 0-7 -> K=8h+i, 8-15 -> K=16+8h+i
    v16bf a[2];
#pragma unroll
    for (int mt = 0; mt < 2; ++mt) {
      const u16t* xr = &Xl[cur][wm * 32 + mt * 16 + ln][0];
      a[mt] = ld_frag(xr + 8 * lh, xr + 16 + 8 * lh);
    }
#pragma unroll
    for (int nt = 0; nt < 4; ++nt) {
      // B fragment: lane ln = col N; elems 0-7 -> K=8h+i, 8-15 -> K=16+8h+i
      const u16t* wr = &Wl[cur][wn * 64 + nt * 16 + ln][8 * lh];
      v16bf b = ld_frag(wr, wr + 16);
#pragma unroll
      for (int mt = 0; mt < 2; ++mt)
        acc[mt][nt] = __builtin_amdgcn_wmma_f32_16x16x32_bf16(
            false, a[mt], false, b, (short)0, acc[mt][nt], false, false);
    }

    if (t < 31) {                      // fill the other buffer
      *(uint4*)&Xl[nxt][lr][lc0 * 8]     = px0;
      *(uint4*)&Xl[nxt][lr][lc0 * 8 + 8] = px1;
      *(uint4*)&Wl[nxt][lr][lc0 * 8]     = pw0;
      *(uint4*)&Wl[nxt][lr][lc0 * 8 + 8] = pw1;
    }
    __syncthreads();
  }

  // C layout: VGPR r, lane -> row = r + 8*(lane/16), col = lane%16
#pragma unroll
  for (int mt = 0; mt < 2; ++mt)
#pragma unroll
    for (int nt = 0; nt < 4; ++nt)
#pragma unroll
      for (int r = 0; r < 8; ++r) {
        int row_g = gm + wm * 32 + mt * 16 + 8 * lh + r;
        int col_g = gn + wn * 64 + nt * 16 + ln;
        if (mode == 0) {
          int bb = row_g >> 10, l = row_g & 1023;
          int h = col_g >> 6, dh = col_g & 63;
          ((u16t*)out)[(((size_t)bb * 16 + h) * 1024 + l) * 64 + dh] =
              f2bf(acc[mt][nt][r]);
        } else {
          ((float*)out)[(size_t)row_g * 1024 + col_g] = acc[mt][nt][r];
        }
      }
}

// ---------------------------------------------------------------------------
// Flash attention: one WG per (b, h, 64-row q block). 4 waves x 16 rows.
// Key blocks of 32; online softmax; q/k/v bf16 [B,H,L,DH]; out bf16 [B,L,D].
// ---------------------------------------------------------------------------
__global__ __launch_bounds__(128) void attn_flash(const u16t* __restrict__ q,
                                                  const u16t* __restrict__ k,
                                                  const u16t* __restrict__ v,
                                                  const float* __restrict__ mask,
                                                  u16t* __restrict__ o) {
  const int rb = blockIdx.x;        // 16 row-blocks of 64
  const int h  = blockIdx.y;        // 16 heads
  const int b  = blockIdx.z;        // 8 batches
  const int tid = threadIdx.x;
  const int wave = tid >> 5, lane = tid & 31;
  const int lh = lane >> 4, ln = lane & 15;

  __shared__ u16t Klds[32][72];       // [key][dh], 144B stride
  __shared__ u16t Vt[64][40];         // [dh][key], 80B stride (transposed V)
  __shared__ u16t Plds[4][16][40];    // per-wave P tile [qrow][key]

  const size_t bh   = (((size_t)b * 16 + h) * 1024) * 64;  // q/k/v base
  const int    row0 = rb * 64 + wave * 16;                 // this wave's rows

  // Q fragments for K-chunks dh 0..31 and 32..63 (loaded once)
  v16bf qa[2];
  {
    const u16t* qrow = q + bh + (size_t)(row0 + ln) * 64;
#pragma unroll
    for (int c = 0; c < 2; ++c)
      qa[c] = ld_frag(qrow + c * 32 + 8 * lh, qrow + c * 32 + 16 + 8 * lh);
  }

  float mrow[8], lsum[8];
  v8f O[4];
#pragma unroll
  for (int r = 0; r < 8; ++r) { mrow[r] = -3.0e38f; lsum[r] = 0.0f; }
#pragma unroll
  for (int nt = 0; nt < 4; ++nt)
#pragma unroll
    for (int i = 0; i < 8; ++i) O[nt][i] = 0.0f;

  const float scale = 0.125f;   // 1/sqrt(64)

  for (int j = 0; j < 1024; j += 32) {
    __syncthreads();   // previous iteration's LDS reads complete
    // Cooperative load: K block [64 keys][32 dh], V transposed into Vt
#pragma unroll
    for (int t = tid; t < 256; t += 128) {
      int row = t >> 2, c4 = t & 3;
      uint4 d = *(const uint4*)(k + bh + (size_t)(j + row) * 64 + c4 * 8);
      *(uint4*)&Klds[row][c4 * 8] = d;
      uint4 e = *(const uint4*)(v + bh + (size_t)(j + row) * 64 + c4 * 8);
      u16t tmp[8];
      *(uint4*)tmp = e;
#pragma unroll
      for (int i = 0; i < 8; ++i) Vt[c4 * 8 + i][row] = tmp[i];
    }
    __syncthreads();

    // S = Q * K^T : two 16x16 subtiles (keys j..j+15, j+16..j+31)
    v8f S[2];
#pragma unroll
    for (int nt = 0; nt < 2; ++nt) {
      v8f s;
#pragma unroll
      for (int i = 0; i < 8; ++i) s[i] = 0.0f;
#pragma unroll
      for (int c = 0; c < 2; ++c) {
        const u16t* kr = &Klds[nt * 16 + ln][c * 32 + 8 * lh];
        v16bf bb = ld_frag(kr, kr + 16);
        s = __builtin_amdgcn_wmma_f32_16x16x32_bf16(
            false, qa[c], false, bb, (short)0, s, false, false);
      }
      S[nt] = s;
    }

    // Mask + scale + online softmax (row = r + 8*lh across 16 lanes)
    float p0a[8], p1a[8], corr[8];
    const int gr = row0 + 8 * lh;
#pragma unroll
    for (int r = 0; r < 8; ++r) {
      const float* mp = mask + (size_t)(gr + r) * 1024 + j;
      float m0 = mp[ln], m1 = mp[16 + ln];
      float s0 = (S[0][r] * m0 - 1.0e9f * (1.0f - m0)) * scale;
      float s1 = (S[1][r] * m1 - 1.0e9f * (1.0f - m1)) * scale;
      float rm = fmaxf(s0, s1);
#pragma unroll
      for (int off = 1; off < 16; off <<= 1)
        rm = fmaxf(rm, __shfl_xor(rm, off, 32));
      float mn = fmaxf(mrow[r], rm);
      float c  = __expf(mrow[r] - mn);
      float p0 = __expf(s0 - mn);
      float p1 = __expf(s1 - mn);
      float rs = p0 + p1;
#pragma unroll
      for (int off = 1; off < 16; off <<= 1) rs += __shfl_xor(rs, off, 32);
      lsum[r] = lsum[r] * c + rs;
      mrow[r] = mn;
      corr[r] = c;
      p0a[r] = p0; p1a[r] = p1;
    }

    // P (C layout) -> LDS -> A layout
#pragma unroll
    for (int r = 0; r < 8; ++r) {
      Plds[wave][r + 8 * lh][ln]      = f2bf(p0a[r]);
      Plds[wave][r + 8 * lh][16 + ln] = f2bf(p1a[r]);
    }
    __syncthreads();

    const u16t* prow = &Plds[wave][ln][0];
    v16bf pa = ld_frag(prow + 8 * lh, prow + 16 + 8 * lh);

    // O = O*corr + P x V  (4 dh-subtiles of 16)
#pragma unroll
    for (int nt = 0; nt < 4; ++nt) {
      v8f a = O[nt];
#pragma unroll
      for (int r = 0; r < 8; ++r) a[r] *= corr[r];
      const u16t* vr = &Vt[nt * 16 + ln][8 * lh];
      v16bf vb = ld_frag(vr, vr + 16);
      O[nt] = __builtin_amdgcn_wmma_f32_16x16x32_bf16(
          false, pa, false, vb, (short)0, a, false, false);
    }
  }

  // Epilogue: normalize and store bf16 into [B, L, D] (d = h*64 + dh)
#pragma unroll
  for (int nt = 0; nt < 4; ++nt)
#pragma unroll
    for (int r = 0; r < 8; ++r) {
      int l  = row0 + 8 * lh + r;
      int dh = nt * 16 + ln;
      float val = O[nt][r] / fmaxf(lsum[r], 1e-30f);
      o[((size_t)b * 1024 + l) * 1024 + h * 64 + dh] = f2bf(val);
    }
}

// ---------------------------------------------------------------------------
extern "C" void kernel_launch(void* const* d_in, const int* in_sizes, int n_in,
                              void* d_out, int out_size, void* d_ws,
                              size_t ws_size, hipStream_t stream) {
  const float* x    = (const float*)d_in[0];
  const float* mask = (const float*)d_in[1];
  const float* Wk   = (const float*)d_in[2];
  const float* Wv   = (const float*)d_in[3];
  const float* Wq   = (const float*)d_in[4];
  const float* Wo   = (const float*)d_in[5];

  const size_t NX = 8ull * 1024 * 1024;   // B*L*D elements
  const size_t NW = 1024ull * 1024;       // D*D elements

  u16t* ws  = (u16t*)d_ws;
  u16t* xb  = ws; ws += NX;
  u16t* wkb = ws; ws += NW;
  u16t* wvb = ws; ws += NW;
  u16t* wqb = ws; ws += NW;
  u16t* wob = ws; ws += NW;
  u16t* qb  = ws; ws += NX;
  u16t* kb  = ws; ws += NX;
  u16t* vb  = ws; ws += NX;
  u16t* ab  = ws; ws += NX;
  if (ws_size < (size_t)((char*)ws - (char*)d_ws)) return;

  int np = (int)(NX / 2);
  cvt_f32_bf16<<<(np + 255) / 256, 256, 0, stream>>>(x, xb, np);
  np = (int)(NW / 2);
  cvt_f32_bf16<<<(np + 255) / 256, 256, 0, stream>>>(Wk, wkb, np);
  cvt_f32_bf16<<<(np + 255) / 256, 256, 0, stream>>>(Wv, wvb, np);
  cvt_f32_bf16<<<(np + 255) / 256, 256, 0, stream>>>(Wq, wqb, np);
  cvt_f32_bf16<<<(np + 255) / 256, 256, 0, stream>>>(Wo, wob, np);

  dim3 ggrid(64, 8);     // 8192/128 x 1024/128
  gemm_bf16<<<ggrid, 256, 0, stream>>>(xb, wqb, (void*)qb, 0);
  gemm_bf16<<<ggrid, 256, 0, stream>>>(xb, wkb, (void*)kb, 0);
  gemm_bf16<<<ggrid, 256, 0, stream>>>(xb, wvb, (void*)vb, 0);

  dim3 agrid(16, 16, 8); // row-blocks x heads x batch
  attn_flash<<<agrid, 128, 0, stream>>>(qb, kb, vb, mask, ab);

  gemm_bf16<<<ggrid, 256, 0, stream>>>(ab, wob, d_out, 1);
}
